// ocvqa_60447369724350
// MI455X (gfx1250) — matrix operand order repaired and checked by reading, not compiled, not run on hardware
//
#include <hip/hip_runtime.h>
#include <cmath>

typedef __attribute__((ext_vector_type(16))) __bf16 v16bf;
typedef __attribute__((ext_vector_type(8)))  __bf16 v8bf;
typedef __attribute__((ext_vector_type(8)))  float  v8f;

#define BATCH 512
#define TQ    20
#define EMBD  128
#define HID   128
#define GATES 512     // 4*HID
#define OBJ   64
#define SLOTS 16
#define RELD  256
#define AVOC  1000

#define RP    264     // padded LDS row stride for 256-wide bf16 tiles (528B -> 4-bank lane advance)
#define HP    136     // padded LDS row stride for 128-wide bf16 tiles (272B)

__device__ __forceinline__ float sigf(float x) { return 1.0f / (1.0f + expf(-x)); }

// Async global->LDS copy of 16 bytes per lane (GLOBAL_LOAD_ASYNC_TO_LDS_B128,
// GVS mode: saddr base + 32-bit vgpr offset; vdst holds the LDS byte offset).
__device__ __forceinline__ void async_copy16(unsigned lds_off, const void* base,
                                             unsigned byte_off) {
    asm volatile("global_load_async_to_lds_b128 %0, %1, %2"
                 :: "v"(lds_off), "v"(byte_off), "s"(base) : "memory");
}
__device__ __forceinline__ void async_wait0() {
    asm volatile("s_wait_asynccnt 0" ::: "memory");
}

// A-matrix (16x32 bf16) fragment (contiguous k order, B loads use the same):
// lane m holds row m; lanes 0-15: K = 0..15 +kbase, lanes 16-31: K = 16..31 +kbase.
__device__ __forceinline__ v16bf a_frag(const __bf16* row, int half, int kbase) {
    v8bf lo = *(const v8bf*)(row + kbase + half * 16);
    v8bf hi = *(const v8bf*)(row + kbase + half * 16 + 8);
    return __builtin_shufflevector(lo, hi, 0, 1, 2, 3, 4, 5, 6, 7,
                                          8, 9, 10, 11, 12, 13, 14, 15);
}

// B-matrix fragment: 16 contiguous bf16 (two 16B LDS loads; 16B-aligned safe).
__device__ __forceinline__ v16bf b_frag(const __bf16* p) {
    v8bf lo = *(const v8bf*)(p);
    v8bf hi = *(const v8bf*)(p + 8);
    return __builtin_shufflevector(lo, hi, 0, 1, 2, 3, 4, 5, 6, 7,
                                          8, 9, 10, 11, 12, 13, 14, 15);
}

// ---------------------------------------------------------------------------
// fp32 [K][N] -> bf16 transposed [N][K]
__global__ void k_cvt_t(const float* __restrict__ src, __bf16* __restrict__ dst,
                        int K, int N) {
    int idx = blockIdx.x * 256 + threadIdx.x;
    if (idx >= K * N) return;
    int n = idx / K, k = idx - n * K;
    dst[idx] = (__bf16)src[k * N + n];
}

// ---------------------------------------------------------------------------
// preZ[r][0:512] = emb[q[r]] @ lstm_k + lstm_b    for r = b*T + t  (10240 rows)
// One block = 16 rows; 8 waves x 4 N-tiles; 4 independent acc chains per wave.
__global__ void k_lstm_pre(const int* __restrict__ q, const float* __restrict__ emb,
                           const __bf16* __restrict__ wkT, const float* __restrict__ bias,
                           float* __restrict__ preZ) {
    __shared__ __bf16 sA[16 * HP];
    int tid = threadIdx.x, blk = blockIdx.x;
    for (int idx = tid; idx < 16 * EMBD; idx += 256) {
        int m = idx >> 7, k = idx & 127;
        int tok = q[blk * 16 + m];               // flat (b,t) row == q flat index
        sA[m * HP + k] = (__bf16)emb[tok * EMBD + k];
    }
    __syncthreads();
    int wave = tid >> 5, lane = tid & 31;
    int lm = lane & 15, half = lane >> 4;
    const __bf16* arow = sA + lm * HP;
    v8f ac[4] = {};
    for (int kk = 0; kk < EMBD; kk += 32) {
        v16bf a = a_frag(arow, half, kk);
#pragma unroll
        for (int u = 0; u < 4; ++u) {
            v16bf b = *(const v16bf*)(wkT + ((wave * 4 + u) * 16 + lm) * EMBD + kk + half * 16);
            ac[u] = __builtin_amdgcn_wmma_f32_16x16x32_bf16(false, a, false, b,
                                                            (short)0, ac[u], false, false);
        }
    }
#pragma unroll
    for (int u = 0; u < 4; ++u) {
        int ncol = (wave * 4 + u) * 16 + lm;
        float bv = bias[ncol];
#pragma unroll
        for (int r = 0; r < 8; ++r) {
            int mr = r + half * 8;
            preZ[(size_t)(blk * 16 + mr) * GATES + ncol] = ac[u][r] + bv;
        }
    }
}

// ---------------------------------------------------------------------------
// Recurrent LSTM: one block = 16 batch rows. Wr^T async-staged once into LDS
// (~136KB), reused for all 20 steps. 4 acc chains per wave per step.
__global__ void k_lstm_rec(const float* __restrict__ preZ, const __bf16* __restrict__ wrT,
                           __bf16* __restrict__ hq) {
    extern __shared__ char dsm[];
    float*  sZ   = (float*)dsm;                         // 16*512 f32   = 32768 B
    float*  sC   = (float*)(dsm + 32768);               // 16*128 f32   =  8192 B
    __bf16* wbuf = (__bf16*)(dsm + 40960);              // 512*HP bf16  = 139264 B
    __bf16* sH   = (__bf16*)(dsm + 40960 + 512 * HP * 2); // 16*HP bf16 =   4352 B
    int tid = threadIdx.x;
    int b0 = blockIdx.x * 16;
    // async-stage Wr^T [512][128] -> LDS padded rows
    for (int i = tid; i < 512 * 16; i += 256) {
        int row = i >> 4, col = i & 15;
        unsigned loff = (unsigned)(uintptr_t)(wbuf + row * HP + col * 8);
        async_copy16(loff, wrT, (unsigned)(i * 16));
    }
    for (int idx = tid; idx < 16 * HID; idx += 256) {
        int m = idx >> 7, k = idx & 127;
        sH[m * HP + k] = (__bf16)0.0f;
        sC[idx] = 0.0f;
    }
    async_wait0();
    __syncthreads();
    int wave = tid >> 5, lane = tid & 31;
    int lm = lane & 15, half = lane >> 4;
    const __bf16* arow = sH + lm * HP;
    for (int t = 0; t < TQ; ++t) {
        v8f ac[4] = {};
        for (int kk = 0; kk < HID; kk += 32) {
            v16bf a = a_frag(arow, half, kk);
#pragma unroll
            for (int u = 0; u < 4; ++u) {
                v16bf b = b_frag(wbuf + ((wave * 4 + u) * 16 + lm) * HP + kk + half * 16);
                ac[u] = __builtin_amdgcn_wmma_f32_16x16x32_bf16(false, a, false, b,
                                                                (short)0, ac[u], false, false);
            }
        }
#pragma unroll
        for (int u = 0; u < 4; ++u) {
            int ncol = (wave * 4 + u) * 16 + lm;
#pragma unroll
            for (int r = 0; r < 8; ++r) {
                int mr = r + half * 8;
                sZ[mr * GATES + ncol] =
                    ac[u][r] + preZ[(size_t)((b0 + mr) * TQ + t) * GATES + ncol];
            }
        }
        __syncthreads();
        for (int idx = tid; idx < 16 * HID; idx += 256) {
            int mr = idx >> 7, h = idx & 127;
            float zi = sZ[mr * GATES + h];
            float zf = sZ[mr * GATES + HID + h];
            float zg = sZ[mr * GATES + 2 * HID + h];
            float zo = sZ[mr * GATES + 3 * HID + h];
            float c = sigf(zf) * sC[idx] + sigf(zi) * tanhf(zg);
            sC[idx] = c;
            sH[mr * HP + h] = (__bf16)(sigf(zo) * tanhf(c));
        }
        __syncthreads();
    }
    for (int idx = tid; idx < 16 * HID; idx += 256) {
        int m = idx >> 7, k = idx & 127;
        hq[(size_t)(b0 + m) * HID + k] = sH[m * HP + k];
    }
}

// ---------------------------------------------------------------------------
// Fused 4-layer rel MLP. Block = (batch b, 64-pair tile). Per layer: async-
// stage the full 256x256 bf16 weight into LDS (conflict-free padded rows),
// prefetch the next layer's weights to L2, then WMMA with 4 independent acc
// chains sharing each A fragment. Deterministic per-block column partials.
__global__ void k_rel(const float* __restrict__ objects, const __bf16* __restrict__ hq,
                      const __bf16* __restrict__ relWT,
                      const float* __restrict__ rb0, const float* __restrict__ rb1,
                      const float* __restrict__ rb2, const float* __restrict__ rb3,
                      float* __restrict__ partial) {
    extern __shared__ char dsm[];
    __bf16* wbuf = (__bf16*)dsm;                 // 256*RP = 135168 B
    __bf16* actA = wbuf + 256 * RP;              //  64*RP =  33792 B
    __bf16* actB = actA + 64 * RP;               //  64*RP =  33792 B
    int tid = threadIdx.x;
    int b = blockIdx.x >> 2, pt = blockIdx.x & 3;
    // gather pair-concat input [o_i | o_j | q] into actA
    for (int idx = tid; idx < 64 * RELD; idx += 256) {
        int pl = idx >> 8, c = idx & 255;
        int p = pt * 64 + pl;
        int i = p >> 4, j = p & 15;
        __bf16 v;
        if (c < 64)        v = (__bf16)objects[((size_t)b * SLOTS + i) * OBJ + c];
        else if (c < 128)  v = (__bf16)objects[((size_t)b * SLOTS + j) * OBJ + (c - 64)];
        else               v = hq[(size_t)b * HID + (c - 128)];
        actA[pl * RP + c] = v;
    }
    int wave = tid >> 5, lane = tid & 31;
    int lm = lane & 15, half = lane >> 4;
    int mt = wave >> 1;                 // all 8 tiles of a wave share the M-tile
    int nt0 = (wave & 1) * 8;
    for (int l = 0; l < 4; ++l) {
        // async-stage this layer's weights [256][256] bf16 -> LDS padded rows
        const void* wsrc = (const void*)(relWT + l * (RELD * RELD));
        for (int i = tid; i < 256 * 32; i += 256) {
            int row = i >> 5, col = i & 31;
            unsigned loff = (unsigned)(uintptr_t)(wbuf + row * RP + col * 8);
            async_copy16(loff, wsrc, (unsigned)(i * 16));
        }
        if (l < 3) {   // prefetch next layer's weights into L2 (global_prefetch_b8)
            const char* nw = (const char*)(relWT + (l + 1) * (RELD * RELD));
            for (int i = tid; i < 512; i += 256) __builtin_prefetch(nw + i * 256, 0, 1);
        }
        async_wait0();
        __syncthreads();
        const __bf16* in  = (l & 1) ? actB : actA;
        __bf16*       out = (l & 1) ? actA : actB;
        const float* bias = (l == 0) ? rb0 : (l == 1) ? rb1 : (l == 2) ? rb2 : rb3;
        const __bf16* arow = in + (mt * 16 + lm) * RP;
        for (int g = 0; g < 2; ++g) {
            int ntb = nt0 + g * 4;
            v8f ac[4] = {};
            for (int kk = 0; kk < RELD; kk += 32) {
                v16bf a = a_frag(arow, half, kk);
#pragma unroll
                for (int u = 0; u < 4; ++u) {
                    v16bf bm = b_frag(wbuf + ((ntb + u) * 16 + lm) * RP + kk + half * 16);
                    ac[u] = __builtin_amdgcn_wmma_f32_16x16x32_bf16(false, a, false, bm,
                                                                    (short)0, ac[u], false, false);
                }
            }
#pragma unroll
            for (int u = 0; u < 4; ++u) {
                int n = (ntb + u) * 16 + lm;
                float bv = bias[n];
#pragma unroll
                for (int r = 0; r < 8; ++r) {
                    int mr = mt * 16 + r + half * 8;
                    float v = ac[u][r] + bv;
                    out[mr * RP + n] = (__bf16)(v > 0.0f ? v : 0.0f);
                }
            }
        }
        __syncthreads();
    }
    // final activations are in actA (layer 3 wrote there)
    float s = 0.0f;
    for (int p = 0; p < 64; ++p) s += (float)actA[p * RP + tid];
    partial[(size_t)blockIdx.x * RELD + tid] = s;
}

// ---------------------------------------------------------------------------
// resp MLP (256->256->256->29) + out (29->1000) + softmax.  One block per batch.
__global__ void k_head(const float* __restrict__ partial,
                       const float* __restrict__ w0, const float* __restrict__ b0v,
                       const float* __restrict__ w1, const float* __restrict__ b1v,
                       const float* __restrict__ w2, const float* __restrict__ b2v,
                       const float* __restrict__ ow, const float* __restrict__ ob,
                       float* __restrict__ outp) {
    __shared__ float sX[256];
    __shared__ float sH[256];
    __shared__ float sH2[32];
    __shared__ float sL[AVOC];
    __shared__ float sR[256];
    int tid = threadIdx.x, b = blockIdx.x;
    {
        float s = 0.0f;
        for (int pt = 0; pt < 4; ++pt) s += partial[((size_t)b * 4 + pt) * RELD + tid];
        sX[tid] = s;
    }
    __syncthreads();
    {
        float acc = b0v[tid];
        for (int k = 0; k < 256; ++k) acc += sX[k] * w0[k * 256 + tid];
        sH[tid] = acc > 0.0f ? acc : 0.0f;
    }
    __syncthreads();
    {
        float acc = b1v[tid];
        for (int k = 0; k < 256; ++k) acc += sH[k] * w1[k * 256 + tid];
        sX[tid] = acc > 0.0f ? acc : 0.0f;
    }
    __syncthreads();
    if (tid < 29) {
        float acc = b2v[tid];
        for (int k = 0; k < 256; ++k) acc += sX[k] * w2[k * 29 + tid];
        sH2[tid] = acc > 0.0f ? acc : 0.0f;
    }
    __syncthreads();
    for (int j = tid; j < AVOC; j += 256) {
        float acc = ob[j];
        for (int k = 0; k < 29; ++k) acc += sH2[k] * ow[k * AVOC + j];
        sL[j] = acc;
    }
    __syncthreads();
    float mx = -3.4e38f;
    for (int j = tid; j < AVOC; j += 256) mx = fmaxf(mx, sL[j]);
    sR[tid] = mx;
    __syncthreads();
    for (int s = 128; s > 0; s >>= 1) {
        if (tid < s) sR[tid] = fmaxf(sR[tid], sR[tid + s]);
        __syncthreads();
    }
    mx = sR[0];
    __syncthreads();
    float sum = 0.0f;
    for (int j = tid; j < AVOC; j += 256) {
        float e = expf(sL[j] - mx);
        sL[j] = e;
        sum += e;
    }
    sR[tid] = sum;
    __syncthreads();
    for (int s = 128; s > 0; s >>= 1) {
        if (tid < s) sR[tid] += sR[tid + s];
        __syncthreads();
    }
    float inv = 1.0f / sR[0];
    for (int j = tid; j < AVOC; j += 256)
        outp[(size_t)b * AVOC + j] = sL[j] * inv;
}

// ---------------------------------------------------------------------------
extern "C" void kernel_launch(void* const* d_in, const int* in_sizes, int n_in,
                              void* d_out, int out_size, void* d_ws, size_t ws_size,
                              hipStream_t stream) {
    const float* objects = (const float*)d_in[0];
    const int*   q       = (const int*)d_in[1];
    const float* emb     = (const float*)d_in[2];
    const float* lstm_k  = (const float*)d_in[3];
    const float* lstm_r  = (const float*)d_in[4];
    const float* lstm_b  = (const float*)d_in[5];
    const float* rel_w[4] = {(const float*)d_in[6], (const float*)d_in[8],
                             (const float*)d_in[10], (const float*)d_in[12]};
    const float* rel_b[4] = {(const float*)d_in[7], (const float*)d_in[9],
                             (const float*)d_in[11], (const float*)d_in[13]};
    const float* resp_w0 = (const float*)d_in[14]; const float* resp_b0 = (const float*)d_in[15];
    const float* resp_w1 = (const float*)d_in[16]; const float* resp_b1 = (const float*)d_in[17];
    const float* resp_w2 = (const float*)d_in[18]; const float* resp_b2 = (const float*)d_in[19];
    const float* out_w   = (const float*)d_in[20]; const float* out_b   = (const float*)d_in[21];

    char* ws = (char*)d_ws;
    __bf16* wkT     = (__bf16*)(ws + 0);            // 512*128 bf16
    __bf16* wrT     = (__bf16*)(ws + 131072);       // 512*128 bf16
    __bf16* relWT   = (__bf16*)(ws + 262144);       // 4*256*256 bf16
    float*  preZ    = (float*)(ws + 786432);        // 10240*512 f32
    __bf16* hq      = (__bf16*)(ws + 21757952);     // 512*128 bf16
    float*  partial = (float*)(ws + 21889024);      // 2048*256 f32

    const int relSmem  = (256 + 64 + 64) * RP * 2;          // 202752 B
    const int lstmSmem = 32768 + 8192 + 512 * HP * 2 + 16 * HP * 2;  // 184576 B
    (void)hipFuncSetAttribute((const void*)k_rel,
                              hipFuncAttributeMaxDynamicSharedMemorySize, relSmem);
    (void)hipFuncSetAttribute((const void*)k_lstm_rec,
                              hipFuncAttributeMaxDynamicSharedMemorySize, lstmSmem);

    k_cvt_t<<<256, 256, 0, stream>>>(lstm_k, wkT, 128, 512);
    k_cvt_t<<<256, 256, 0, stream>>>(lstm_r, wrT, 128, 512);
    for (int l = 0; l < 4; ++l)
        k_cvt_t<<<256, 256, 0, stream>>>(rel_w[l], relWT + l * 65536, 256, 256);

    k_lstm_pre<<<640, 256, 0, stream>>>(q, emb, wkT, lstm_b, preZ);
    k_lstm_rec<<<32, 256, lstmSmem, stream>>>(preZ, wrT, hq);
    k_rel<<<2048, 256, relSmem, stream>>>(objects, hq, relWT,
                                          rel_b[0], rel_b[1], rel_b[2], rel_b[3], partial);
    k_head<<<512, 256, 0, stream>>>(partial, resp_w0, resp_b0, resp_w1, resp_b1,
                                    resp_w2, resp_b2, out_w, out_b, (float*)d_out);
}
